// CausalAttention_8375186227780
// MI455X (gfx1250) — compile-verified, matrix operand order checked by reading, not run on hardware
//
#include <hip/hip_runtime.h>

typedef __bf16 bf16;
typedef __attribute__((ext_vector_type(8)))  bf16  bf16x8;
typedef __attribute__((ext_vector_type(16))) bf16  bf16x16;
typedef __attribute__((ext_vector_type(8)))  float f32x8;

#define LN_ROPE_BASE 10.81977828441028f   /* ln(50000) */
#define SCALE_QK     0.08838834764831845f /* 1/sqrt(128) */

#if defined(__HIP_DEVICE_COMPILE__) && defined(__gfx1250__)
#define CDNA5_ASYNC 1
#else
#define CDNA5_ASYNC 0
#endif

// ---------------------------------------------------------------------------
// CDNA5 async global->LDS copy (ASYNCcnt-tracked). LDS byte offset is the low
// 32 bits of the generic pointer (ISA: LDS aperture keeps offset in addr[31:0]).
// ---------------------------------------------------------------------------
__device__ __forceinline__ void async_b128(const void* g, void* l) {
#if CDNA5_ASYNC
  unsigned loff = (unsigned)(uintptr_t)l;
  unsigned long long ga = (unsigned long long)(uintptr_t)g;
  asm volatile("global_load_async_to_lds_b128 %0, %1, off"
               :: "v"(loff), "v"(ga) : "memory");
#else
  *(float4*)l = *(const float4*)g;
#endif
}

template <int N>
__device__ __forceinline__ void async_wait() {
#if CDNA5_ASYNC
  asm volatile("s_wait_asynccnt %0" :: "i"(N) : "memory");
#endif
}

// ---------------------------------------------------------------------------
// WMMA fragment helpers (16x16x32 bf16, wave32)
// A per-lane: row = lane%16; elems 0..7  = K[khalf*8 .. +8)
//             elems 8..15 = K[16+khalf*8 .. +8)       (khalf = lane/16)
// B operand is staged N-major so it uses the identical pattern.
// C/D: VGPR r holds row (r + 8*(lane/16)), col = lane%16.
// ---------------------------------------------------------------------------
__device__ __forceinline__ bf16x16 ldsFrag(const bf16* rowPtr, int khalf) {
  bf16x8 lo = *(const bf16x8*)(rowPtr + khalf * 8);
  bf16x8 hi = *(const bf16x8*)(rowPtr + 16 + khalf * 8);
  return __builtin_shufflevector(lo, hi, 0,1,2,3,4,5,6,7,8,9,10,11,12,13,14,15);
}

__device__ __forceinline__ f32x8 wmma_bf16(bf16x16 a, bf16x16 b, f32x8 c) {
  return __builtin_amdgcn_wmma_f32_16x16x32_bf16(false, a, false, b, (short)0, c,
                                                 false, false);
}

// ---------------------------------------------------------------------------
// Elementwise f32 -> bf16
// ---------------------------------------------------------------------------
__global__ __launch_bounds__(256) void cvt_f32_bf16(const float* __restrict__ in,
                                                    bf16* __restrict__ out, int n) {
  int i = blockIdx.x * 256 + threadIdx.x;
  if (i < n) out[i] = (bf16)in[i];
}

// ---------------------------------------------------------------------------
// Tiled transpose + convert: out[c][r] = (bf16)in[r][c], in is [R][C] f32.
// block = 256 (32x8), tile 32x32, grid = (C/32, R/32).
// ---------------------------------------------------------------------------
__global__ __launch_bounds__(256) void cvt_transpose(const float* __restrict__ in,
                                                     bf16* __restrict__ out,
                                                     int R, int C) {
  __shared__ float tl[32][33];
  const int bx = blockIdx.x * 32;  // col base (in)
  const int by = blockIdx.y * 32;  // row base (in)
  const int tx = threadIdx.x & 31, ty = threadIdx.x >> 5;
#pragma unroll
  for (int i = 0; i < 32; i += 8)
    tl[ty + i][tx] = in[(size_t)(by + ty + i) * C + bx + tx];
  __syncthreads();
#pragma unroll
  for (int i = 0; i < 32; i += 8)
    out[(size_t)(bx + ty + i) * R + by + tx] = (bf16)tl[tx][ty + i];
}

// ---------------------------------------------------------------------------
// Per-head V transpose: Vf [4096][2048] f32 -> Vt [(b*16+h)*128 + dh][2048] bf16
// grid = (2048/32, 128/32, 32), block 256 (32x8).
// ---------------------------------------------------------------------------
__global__ __launch_bounds__(256) void cvt_transpose_v(const float* __restrict__ in,
                                                       bf16* __restrict__ out) {
  __shared__ float tl[32][33];
  const int bh = blockIdx.z, b = bh >> 4, h = bh & 15;
  const int t0 = blockIdx.x * 32, d0 = blockIdx.y * 32;
  const int tx = threadIdx.x & 31, ty = threadIdx.x >> 5;
#pragma unroll
  for (int i = 0; i < 32; i += 8)
    tl[ty + i][tx] = in[(size_t)(b * 2048 + t0 + ty + i) * 2048 + h * 128 + d0 + tx];
  __syncthreads();
#pragma unroll
  for (int i = 0; i < 32; i += 8)
    out[((size_t)bh * 128 + d0 + ty + i) * 2048 + t0 + tx] = (bf16)tl[tx][ty + i];
}

// ---------------------------------------------------------------------------
// C[M,N](f32) = A[M,K](bf16,row-major) * Bt[N,K](bf16, pre-transposed weights).
// 128x128 tiles, k-step 32, double-buffered async global->LDS staging with a
// peeled (branch-free) steady-state loop. 8 waves (4x2); 2x4 frags per wave.
// ---------------------------------------------------------------------------
__global__ __launch_bounds__(256) void gemm_bf16(const bf16* __restrict__ A,
                                                 const bf16* __restrict__ Bt,
                                                 float* __restrict__ C,
                                                 int M, int N, int K) {
  __shared__ bf16 As[2][128][40];  // [buf][m][k] (stride 40 -> 16B-aligned chunks)
  __shared__ bf16 Bs[2][128][40];  // [buf][n][k]

  const int tid  = threadIdx.x;
  const int lane = tid & 31, wave = tid >> 5;
  const int wr = wave >> 1, wc = wave & 1;
  const int hlf = lane >> 4, l16 = lane & 15;
  const int mbase = blockIdx.x * 128;
  const int nbase = blockIdx.y * 128;
  const size_t Ks = (size_t)K, Ns = (size_t)N;

  const bf16* Arow = A + (size_t)(mbase + (tid >> 1)) * Ks;
  const bf16* Brow = Bt + (size_t)(nbase + (tid >> 1)) * Ks;
  const int ar = tid >> 1, ac = (tid & 1) * 16;

  f32x8 acc[2][4];
#pragma unroll
  for (int i = 0; i < 2; i++)
#pragma unroll
    for (int j = 0; j < 4; j++) acc[i][j] = {};

  auto issue = [&](int kb, int sel) {  // 4 async b128 per thread per tile
    async_b128(Arow + kb + ac,     &As[sel][ar][ac]);
    async_b128(Arow + kb + ac + 8, &As[sel][ar][ac + 8]);
    async_b128(Brow + kb + ac,     &Bs[sel][ar][ac]);
    async_b128(Brow + kb + ac + 8, &Bs[sel][ar][ac + 8]);
  };

  auto compute = [&](int cur) {
    bf16x16 af[2], bfr[4];
#pragma unroll
    for (int i = 0; i < 2; i++)
      af[i] = ldsFrag(&As[cur][wr * 32 + i * 16 + l16][0], hlf);
#pragma unroll
    for (int j = 0; j < 4; j++)
      bfr[j] = ldsFrag(&Bs[cur][wc * 64 + j * 16 + l16][0], hlf);
#pragma unroll
    for (int i = 0; i < 2; i++)
#pragma unroll
      for (int j = 0; j < 4; j++) acc[i][j] = wmma_bf16(af[i], bfr[j], acc[i][j]);
  };

  issue(0, 0);
  int kb = 0;
  for (; kb < K - 32; kb += 32) {           // branch-free steady state
    const int cur = (kb >> 5) & 1;
    issue(kb + 32, cur ^ 1);                // prefetch next tile
    async_wait<4>();                        // in-order: current tile landed
    __syncthreads();
    compute(cur);
    __syncthreads();                        // buffer reuse fence
  }
  {                                         // peeled last tile
    const int cur = (kb >> 5) & 1;
    async_wait<0>();
    __syncthreads();
    compute(cur);
  }

#pragma unroll
  for (int i = 0; i < 2; i++)
#pragma unroll
    for (int j = 0; j < 4; j++) {
      const int row0 = mbase + wr * 32 + i * 16 + hlf * 8;
      const int col  = nbase + wc * 64 + j * 16 + l16;
#pragma unroll
      for (int r = 0; r < 8; r++) C[(size_t)(row0 + r) * Ns + col] = acc[i][j][r];
    }
}

// ---------------------------------------------------------------------------
// RMSNorm over inner dim (2048) + full-head RoPE, f32 in -> bf16 out.
// outscale folds the attention 1/sqrt(Dh) into Q at no cost.
// ---------------------------------------------------------------------------
__global__ __launch_bounds__(256) void rmsnorm_rope(const float* __restrict__ X,
                                                    const float* __restrict__ W,
                                                    bf16* __restrict__ Y,
                                                    float outscale) {
  __shared__ float rowbuf[2048];
  __shared__ float red[8];
  const int r = blockIdx.x, tid = threadIdx.x;
  const float* xr = X + (size_t)r * 2048;

  float ss = 0.f;
#pragma unroll
  for (int e = 0; e < 8; e++) {
    int c = tid + e * 256;
    float v = xr[c];
    rowbuf[c] = v;
    ss += v * v;
  }
#pragma unroll
  for (int o = 16; o > 0; o >>= 1) ss += __shfl_xor(ss, o, 32);
  if ((tid & 31) == 0) red[tid >> 5] = ss;
  __syncthreads();
  float tot = red[0];
#pragma unroll
  for (int w = 1; w < 8; w++) tot += red[w];
  const float inv = rsqrtf(tot * (1.0f / 2048.0f) + 1e-6f);
  const int pos = r & 2047;

#pragma unroll
  for (int e = 0; e < 8; e++) {
    int c  = tid + e * 256;
    int dh = c & 127, j = dh & 63;
    float v   = rowbuf[c] * inv * W[c];
    int pc    = (dh < 64) ? c + 64 : c - 64;
    float pv  = rowbuf[pc] * inv * W[pc];
    float rot = (dh < 64) ? -pv : pv;
    float ang = (float)pos * __expf(-(float)j * 0.015625f * LN_ROPE_BASE);
    float sn, cs;
    __sincosf(ang, &sn, &cs);
    Y[(size_t)r * 2048 + c] = (bf16)((v * cs + rot * sn) * outscale);
  }
}

// ---------------------------------------------------------------------------
// Causal flash attention (WMMA). Q (pre-scaled by 1/sqrt(Dh)), K: [4096][2048]
// bf16 (row = b*2048+t, col = h*128+dh). Vt: per-head transposed
// [(b*16+h)*128 + dh][2048] bf16. Block = 256 threads (8 waves): one (b,h),
// 128 query rows; wave w owns q rows [w*16, w*16+16). Async staging; V^T fetch
// overlaps S = Q*K^T. Dynamic LDS = 4 * 128 * 136 * 2 bytes.
// ---------------------------------------------------------------------------
__global__ __launch_bounds__(256) void flash_attn(const bf16* __restrict__ Q,
                                                  const bf16* __restrict__ Km,
                                                  const bf16* __restrict__ Vt,
                                                  bf16* __restrict__ O) {
  extern __shared__ char smem[];
  bf16(*Qs)[136] = (bf16(*)[136])smem;
  bf16(*Ks)[136] = (bf16(*)[136])(smem + 1 * 128 * 136 * sizeof(bf16));
  bf16(*Vs)[136] = (bf16(*)[136])(smem + 2 * 128 * 136 * sizeof(bf16));
  bf16(*Ps)[136] = (bf16(*)[136])(smem + 3 * 128 * 136 * sizeof(bf16));

  const int tid = threadIdx.x, lane = tid & 31, wave = tid >> 5;
  const int hlf = lane >> 4, l16 = lane & 15;
  const int qt = blockIdx.x, bh = blockIdx.y;
  const int b = bh >> 4, h = bh & 15;
  const size_t rowbase = (size_t)b * 2048;
  const int coff = h * 128;
  const int q0 = qt * 128;
  const int rr = tid >> 1, c0 = (tid & 1) * 64;  // staging: row, 64-elem half

  {  // stage Q tile (async, completes at first wait)
    const bf16* src = Q + (rowbase + q0 + rr) * 2048 + coff + c0;
#pragma unroll
    for (int ch = 0; ch < 8; ch++) async_b128(src + ch * 8, &Qs[rr][c0 + ch * 8]);
  }

  f32x8 oacc[8];
  float m_i[8], l_i[8];
#pragma unroll
  for (int j = 0; j < 8; j++) oacc[j] = {};
#pragma unroll
  for (int r = 0; r < 8; r++) { m_i[r] = -__builtin_inff(); l_i[r] = 0.f; }

  const int myq0 = q0 + wave * 16;

  for (int kt = 0; kt <= qt; kt++) {
    const int k0 = kt * 128;
    {  // stage K tile [key][dh] (Ks free: all S-reads done before prev barrier)
      const bf16* src = Km + (rowbase + k0 + rr) * 2048 + coff + c0;
#pragma unroll
      for (int ch = 0; ch < 8; ch++) async_b128(src + ch * 8, &Ks[rr][c0 + ch * 8]);
    }
    async_wait<0>();
    __syncthreads();  // K (and Q) visible to all; everyone past prev O-compute

    {  // stage V^T tile [dh][key] -- overlaps with S compute below
      const bf16* src = Vt + ((size_t)bh * 128 + rr) * 2048 + k0 + c0;
#pragma unroll
      for (int ch = 0; ch < 8; ch++) async_b128(src + ch * 8, &Vs[rr][c0 + ch * 8]);
    }

    // S = Q * K^T  (K tile is naturally N-major for the B operand)
    bf16x16 aq[4];
#pragma unroll
    for (int ks = 0; ks < 4; ks++) aq[ks] = ldsFrag(&Qs[wave * 16 + l16][ks * 32], hlf);
    f32x8 sf[8];
#pragma unroll
    for (int j = 0; j < 8; j++) {
      f32x8 s = {};
#pragma unroll
      for (int ks = 0; ks < 4; ks++)
        s = wmma_bf16(aq[ks], ldsFrag(&Ks[j * 16 + l16][ks * 32], hlf), s);
      sf[j] = s;
    }

    // causal mask + online softmax (scale pre-folded into Q)
#pragma unroll
    for (int r = 0; r < 8; r++) {
      const int qrow = myq0 + 8 * hlf + r;
      float mx = m_i[r];
#pragma unroll
      for (int j = 0; j < 8; j++) {
        int kcol = k0 + j * 16 + l16;
        float v = (kcol <= qrow) ? sf[j][r] : -3.0e38f;
        sf[j][r] = v;
        mx = fmaxf(mx, v);
      }
#pragma unroll
      for (int o = 1; o < 16; o <<= 1) mx = fmaxf(mx, __shfl_xor(mx, o, 32));
      const float corr = __expf(m_i[r] - mx);
      float lsum = 0.f;
#pragma unroll
      for (int j = 0; j < 8; j++) {
        float pp = __expf(sf[j][r] - mx);
        sf[j][r] = pp;
        lsum += pp;
      }
#pragma unroll
      for (int o = 1; o < 16; o <<= 1) lsum += __shfl_xor(lsum, o, 32);
      l_i[r] = l_i[r] * corr + lsum;
      m_i[r] = mx;
#pragma unroll
      for (int j = 0; j < 8; j++) oacc[j][r] *= corr;
    }

    // P: C-layout -> wave-private LDS rows (bf16); own-wave DS ops stay in order
#pragma unroll
    for (int r = 0; r < 8; r++)
#pragma unroll
      for (int j = 0; j < 8; j++)
        Ps[wave * 16 + 8 * hlf + r][j * 16 + l16] = (bf16)sf[j][r];

    async_wait<0>();  // V^T landed
    __syncthreads();

    // O += P * V
    bf16x16 ap[4];
#pragma unroll
    for (int ks = 0; ks < 4; ks++) ap[ks] = ldsFrag(&Ps[wave * 16 + l16][ks * 32], hlf);
#pragma unroll
    for (int j = 0; j < 8; j++)
#pragma unroll
      for (int ks = 0; ks < 4; ks++)
        oacc[j] = wmma_bf16(ap[ks], ldsFrag(&Vs[j * 16 + l16][ks * 32], hlf), oacc[j]);
  }

  // epilogue: divide by l, store bf16
#pragma unroll
  for (int r = 0; r < 8; r++) {
    const int qrow = myq0 + 8 * hlf + r;
    const float linv = 1.0f / l_i[r];
#pragma unroll
    for (int j = 0; j < 8; j++)
      O[(rowbase + qrow) * 2048 + coff + j * 16 + l16] = (bf16)(oacc[j][r] * linv);
  }
}

// ---------------------------------------------------------------------------
extern "C" void kernel_launch(void* const* d_in, const int* in_sizes, int n_in,
                              void* d_out, int out_size, void* d_ws, size_t ws_size,
                              hipStream_t stream) {
  (void)in_sizes; (void)n_in; (void)out_size; (void)ws_size;
  const float* x  = (const float*)d_in[0];
  const float* Wq = (const float*)d_in[1];
  const float* Wk = (const float*)d_in[2];
  const float* Wv = (const float*)d_in[3];
  const float* Wo = (const float*)d_in[4];
  const float* qn = (const float*)d_in[5];
  const float* kn = (const float*)d_in[6];
  float* out = (float*)d_out;

  const int ROWS = 4096;      // B*n
  const int D    = 2048;      // d == inner
  const int NX   = ROWS * D;
  const int NW   = D * D;

  char* p = (char*)d_ws;
  auto carve = [&](size_t bytes) {
    char* r = p;
    p += (bytes + 255) & ~(size_t)255;
    return r;
  };
  bf16*  xb  = (bf16*)carve((size_t)NX * 2);
  bf16*  Wqt = (bf16*)carve((size_t)NW * 2);   // transposed [N][K]
  bf16*  Wkt = (bf16*)carve((size_t)NW * 2);
  bf16*  Wvt = (bf16*)carve((size_t)NW * 2);
  bf16*  Wot = (bf16*)carve((size_t)NW * 2);
  float* Qf  = (float*)carve((size_t)NX * 4);
  float* Kf  = (float*)carve((size_t)NX * 4);
  float* Vf  = (float*)carve((size_t)NX * 4);
  bf16*  Qb  = (bf16*)carve((size_t)NX * 2);
  bf16*  Kb  = (bf16*)carve((size_t)NX * 2);
  bf16*  Vtb = (bf16*)carve((size_t)NX * 2);   // per-head transposed V
  bf16*  AOb = (bf16*)carve((size_t)NX * 2);

  dim3 blk(256);

  // 1) precision conversion; weights transposed to [N][K] for async staging
  cvt_f32_bf16<<<dim3((NX + 255) / 256), blk, 0, stream>>>(x, xb, NX);
  dim3 tgrid(D / 32, D / 32);
  cvt_transpose<<<tgrid, blk, 0, stream>>>(Wq, Wqt, D, D);
  cvt_transpose<<<tgrid, blk, 0, stream>>>(Wk, Wkt, D, D);
  cvt_transpose<<<tgrid, blk, 0, stream>>>(Wv, Wvt, D, D);
  cvt_transpose<<<tgrid, blk, 0, stream>>>(Wo, Wot, D, D);

  // 2) QKV projections (WMMA GEMMs, double-buffered async staging)
  dim3 ggrid(ROWS / 128, D / 128);
  gemm_bf16<<<ggrid, blk, 0, stream>>>(xb, Wqt, Qf, ROWS, D, D);
  gemm_bf16<<<ggrid, blk, 0, stream>>>(xb, Wkt, Kf, ROWS, D, D);
  gemm_bf16<<<ggrid, blk, 0, stream>>>(xb, Wvt, Vf, ROWS, D, D);

  // 3) RMSNorm + RoPE -> bf16 Q/K (Q pre-scaled by 1/sqrt(Dh)); V transpose
  rmsnorm_rope<<<dim3(ROWS), blk, 0, stream>>>(Qf, qn, Qb, SCALE_QK);
  rmsnorm_rope<<<dim3(ROWS), blk, 0, stream>>>(Kf, kn, Kb, 1.0f);
  cvt_transpose_v<<<dim3(2048 / 32, 128 / 32, 32), blk, 0, stream>>>(Vf, Vtb);

  // 4) causal flash attention (WMMA + async staging)
  size_t shmem = (size_t)4 * 128 * 136 * sizeof(bf16);  // 139,264 B of 320 KB WGP LDS
  flash_attn<<<dim3(2048 / 128, 32), blk, shmem, stream>>>(Qb, Kb, Vtb, AOb);

  // 5) output projection -> f32 d_out
  gemm_bf16<<<ggrid, blk, 0, stream>>>(AOb, Wot, out, ROWS, D, D);
}
